// ShallowRegressionLSTM_44487271252615
// MI455X (gfx1250) — matrix operand order repaired and checked by reading, not compiled, not run on hardware
//
#include <hip/hip_runtime.h>

// ---------------- problem constants ----------------
#define Bsz 16384
#define Hh  128
#define Tt  96
#define Ll  24
#define Gg  512          // 4*H
#define BT  32           // batch rows per workgroup
#define XW  119          // L-1+T
#define GST 516          // gates LDS row stride (dwords)
#define KC1 7            // layer1 K chunks: 96/32 + 128/32
#define KC2 8            // layer2 K chunks: 128/32 + 128/32
#define NT  32           // N tiles (512/16)

typedef unsigned int u32;
typedef __bf16 bf16_t;
typedef bf16_t v16bf __attribute__((ext_vector_type(16)));
typedef float  v8f   __attribute__((ext_vector_type(8)));
typedef u32    v8u   __attribute__((ext_vector_type(8)));

// fragment counts (dwords)
#define FR1 (NT * KC1 * 256)   // 57344 dwords per layer-1 buffer
#define FR2 (NT * KC2 * 256)   // 65536 dwords per layer-2 buffer

// d_out offsets
#define OSEQ 0
#define OH1  (Bsz * Ll)
#define OC1  (OH1 + Bsz * Hh)
#define OH2  (OC1 + Bsz * Hh)
#define OC2  (OH2 + Bsz * Hh)

// ---------------- helpers ----------------
__device__ __forceinline__ u32 bf16_hi_bits(float v) {
    u32 b = __builtin_bit_cast(u32, v);
    u32 r = b + 0x7FFFu + ((b >> 16) & 1u);   // round-to-nearest-even
    return r >> 16;
}
__device__ __forceinline__ float bf16_bits_to_f(u32 hb) {
    return __builtin_bit_cast(float, hb << 16);
}
__device__ __forceinline__ void split2(float v, u32& hb, u32& lb) {
    hb = bf16_hi_bits(v);
    float lo = v - bf16_bits_to_f(hb);
    lb = bf16_hi_bits(lo);
}
__device__ __forceinline__ float sigf(float v) {
    return 1.0f / (1.0f + __expf(-v));
}
__device__ __forceinline__ float tanhf_fast(float v) {
    return 1.0f - 2.0f / (1.0f + __expf(2.0f * v));
}

// A fragment (16x32 bf16) from an LDS pack buffer of bf16-pairs.
// row already includes mt*16 + (lane&15); pb = pair base (16*chunk); kh = lane>>4.
__device__ __forceinline__ v16bf lda(const u32* base, int stride, int row, int pb, int kh) {
    const u32* p = base + row * stride + pb + kh * 8;
    v8u a;
    a[0] = p[0]; a[1] = p[1]; a[2] = p[2]; a[3] = p[3];
    a[4] = p[4]; a[5] = p[5]; a[6] = p[6]; a[7] = p[7];
    return __builtin_bit_cast(v16bf, a);
}

// B fragment (32x16 bf16) from pre-swizzled global buffer: 256 dwords per fragment,
// lane-major (8 consecutive dwords per lane).
__device__ __forceinline__ v16bf ldb(const u32* __restrict__ ws, int fragIdx, int lane) {
    const uint4* p = (const uint4*)(ws + (fragIdx << 8) + (lane << 3));
    uint4 q0 = p[0], q1 = p[1];
    v8u b;
    b[0] = q0.x; b[1] = q0.y; b[2] = q0.z; b[3] = q0.w;
    b[4] = q1.x; b[5] = q1.y; b[6] = q1.z; b[7] = q1.w;
    return __builtin_bit_cast(v16bf, b);
}

// ---------------- weight prep: f32 -> hi/lo bf16 WMMA fragments ----------------
__global__ void prep_weights(const float* __restrict__ Wih1, const float* __restrict__ Whh1,
                             const float* __restrict__ Wih2, const float* __restrict__ Whh2,
                             u32* __restrict__ wsH1, u32* __restrict__ wsL1,
                             u32* __restrict__ wsH2, u32* __restrict__ wsL2) {
    int idx = blockIdx.x * blockDim.x + threadIdx.x;
    if (idx < FR1) {
        int nt   = idx / (KC1 * 256);
        int rem  = idx % (KC1 * 256);
        int c    = rem >> 8;
        int q    = rem & 255;
        int lane = q >> 3, v = q & 7;
        int n = nt * 16 + (lane & 15);
        int k = c * 32 + (lane >> 4) * 16 + 2 * v;         // K index into [x(96); h1(128)]
        float w0 = (k   < Tt) ? Wih1[n * Tt + k]       : Whh1[n * Hh + (k - Tt)];
        float w1 = (k+1 < Tt) ? Wih1[n * Tt + (k + 1)] : Whh1[n * Hh + (k + 1 - Tt)];
        u32 h0, l0, h1b, l1b;
        split2(w0, h0, l0);
        split2(w1, h1b, l1b);
        wsH1[idx] = h0 | (h1b << 16);
        wsL1[idx] = l0 | (l1b << 16);
    } else if (idx < FR1 + FR2) {
        int id2  = idx - FR1;
        int nt   = id2 / (KC2 * 256);
        int rem  = id2 % (KC2 * 256);
        int c    = rem >> 8;
        int q    = rem & 255;
        int lane = q >> 3, v = q & 7;
        int n = nt * 16 + (lane & 15);
        int k = c * 32 + (lane >> 4) * 16 + 2 * v;         // K index into [h1(128); h2(128)]
        float w0 = (k   < Hh) ? Wih2[n * Hh + k]       : Whh2[n * Hh + (k - Hh)];
        float w1 = (k+1 < Hh) ? Wih2[n * Hh + (k + 1)] : Whh2[n * Hh + (k + 1 - Hh)];
        u32 h0, l0, h1b, l1b;
        split2(w0, h0, l0);
        split2(w1, h1b, l1b);
        wsH2[id2] = h0 | (h1b << 16);
        wsL2[id2] = l0 | (l1b << 16);
    }
}

// ---------------- main persistent LSTM kernel ----------------
// LDS layout (dwords):
#define SM_XT    0                    // float [32][120]
#define SM_XPH   (SM_XT + BT*120)     // u32   [32][49]
#define SM_XPL   (SM_XPH + BT*49)
#define SM_H1H   (SM_XPL + BT*49)     // u32   [32][65]
#define SM_H1L   (SM_H1H + BT*65)
#define SM_H2H   (SM_H1L + BT*65)
#define SM_H2L   (SM_H2H + BT*65)
#define SM_GATES (SM_H2L + BT*65)     // float [32][516]
#define SM_C1    (SM_GATES + BT*GST)  // float [32][128]
#define SM_C2    (SM_C1 + BT*Hh)
#define SM_BS1   (SM_C2 + BT*Hh)      // float [512]
#define SM_BS2   (SM_BS1 + Gg)
#define SM_WFC   (SM_BS2 + Gg)        // float [128]
#define SM_YRED  (SM_WFC + Hh)        // float [32][8]
#define SM_TOTAL (SM_YRED + BT*8)     // dwords
#define SMEM_BYTES (SM_TOTAL * 4)

__launch_bounds__(256, 1)
__global__ void lstm_main(const float* __restrict__ x,
                          const float* __restrict__ h1_0, const float* __restrict__ c1_0,
                          const float* __restrict__ h2_0, const float* __restrict__ c2_0,
                          const float* __restrict__ bih1, const float* __restrict__ bhh1,
                          const float* __restrict__ bih2, const float* __restrict__ bhh2,
                          const float* __restrict__ wfc,  const float* __restrict__ bfc,
                          const u32* __restrict__ wsH1, const u32* __restrict__ wsL1,
                          const u32* __restrict__ wsH2, const u32* __restrict__ wsL2,
                          float* __restrict__ out) {
    extern __shared__ char smem_raw[];
    float* xt    = (float*)smem_raw + SM_XT;
    u32*   xpH   = (u32*)smem_raw + SM_XPH;
    u32*   xpL   = (u32*)smem_raw + SM_XPL;
    u32*   h1H   = (u32*)smem_raw + SM_H1H;
    u32*   h1L   = (u32*)smem_raw + SM_H1L;
    u32*   h2H   = (u32*)smem_raw + SM_H2H;
    u32*   h2L   = (u32*)smem_raw + SM_H2L;
    float* gates = (float*)smem_raw + SM_GATES;
    float* cb1   = (float*)smem_raw + SM_C1;
    float* cb2   = (float*)smem_raw + SM_C2;
    float* bs1   = (float*)smem_raw + SM_BS1;
    float* bs2   = (float*)smem_raw + SM_BS2;
    float* wfcs  = (float*)smem_raw + SM_WFC;
    float* yred  = (float*)smem_raw + SM_YRED;

    const int tid  = threadIdx.x;
    const int row0 = blockIdx.x * BT;

    // wave/lane geometry for WMMA phases
    const int lane = tid & 31;
    const int w    = tid >> 5;       // 0..7
    const int mt   = w & 1;          // M tile (2 of them)
    const int ng   = w >> 1;         // N group (4 groups of 8 N-tiles)
    const int ln   = lane & 15;
    const int kh   = lane >> 4;
    const int arow = mt * 16 + ln;   // A row within batch tile

    // ---- init: x tile, bias sums, fc weights, c state, h packs ----
    for (int i = tid; i < BT * XW; i += 256) {
        int r = i / XW, c = i % XW;
        xt[r * 120 + c] = x[(size_t)(row0 + r) * XW + c];
    }
    for (int j = tid; j < Gg; j += 256) {
        bs1[j] = bih1[j] + bhh1[j];
        bs2[j] = bih2[j] + bhh2[j];
    }
    if (tid < Hh) wfcs[tid] = wfc[tid];
    for (int i = tid; i < BT * Hh; i += 256) {
        int r = i / Hh, j = i % Hh;
        cb1[r * Hh + j] = c1_0[(size_t)(row0 + r) * Hh + j];
        cb2[r * Hh + j] = c2_0[(size_t)(row0 + r) * Hh + j];
    }
    for (int i = tid; i < BT * 64; i += 256) {
        int r = i / 64, p = i % 64;
        float a0 = h1_0[(size_t)(row0 + r) * Hh + 2 * p];
        float a1 = h1_0[(size_t)(row0 + r) * Hh + 2 * p + 1];
        u32 hb0, lb0, hb1, lb1;
        split2(a0, hb0, lb0); split2(a1, hb1, lb1);
        h1H[r * 65 + p] = hb0 | (hb1 << 16);
        h1L[r * 65 + p] = lb0 | (lb1 << 16);
        a0 = h2_0[(size_t)(row0 + r) * Hh + 2 * p];
        a1 = h2_0[(size_t)(row0 + r) * Hh + 2 * p + 1];
        split2(a0, hb0, lb0); split2(a1, hb1, lb1);
        h2H[r * 65 + p] = hb0 | (hb1 << 16);
        h2L[r * 65 + p] = lb0 | (lb1 << 16);
    }
    __syncthreads();

    for (int t = 0; t < Ll; ++t) {
        // ---- (a) pack x window [32 rows x 96] into hi/lo bf16 pairs ----
        for (int i = tid; i < BT * 48; i += 256) {
            int r = i / 48, p = i % 48;
            float a0 = xt[r * 120 + t + 2 * p];
            float a1 = xt[r * 120 + t + 2 * p + 1];
            u32 hb0, lb0, hb1, lb1;
            split2(a0, hb0, lb0); split2(a1, hb1, lb1);
            xpH[r * 49 + p] = hb0 | (hb1 << 16);
            xpL[r * 49 + p] = lb0 | (lb1 << 16);
        }
        __syncthreads();

        // ---- (b) layer-1 gates = [x|h1] x W1cat^T  (hi/lo split, 3 WMMAs/tile) ----
        for (int n2 = 0; n2 < 8; ++n2) {
            int nt = ng * 8 + n2;
            v8f acc = {};
            #pragma unroll
            for (int c = 0; c < KC1; ++c) {
                v16bf aH, aL;
                if (c < 3) {
                    aH = lda(xpH, 49, arow, 16 * c, kh);
                    aL = lda(xpL, 49, arow, 16 * c, kh);
                } else {
                    aH = lda(h1H, 65, arow, 16 * (c - 3), kh);
                    aL = lda(h1L, 65, arow, 16 * (c - 3), kh);
                }
                int fi = nt * KC1 + c;
                v16bf bH = ldb(wsH1, fi, lane);
                v16bf bL = ldb(wsL1, fi, lane);
                acc = __builtin_amdgcn_wmma_f32_16x16x32_bf16(false, aH, false, bH, (short)0, acc, false, false);
                acc = __builtin_amdgcn_wmma_f32_16x16x32_bf16(false, aH, false, bL, (short)0, acc, false, false);
                acc = __builtin_amdgcn_wmma_f32_16x16x32_bf16(false, aL, false, bH, (short)0, acc, false, false);
            }
            #pragma unroll
            for (int v = 0; v < 8; ++v)
                gates[(mt * 16 + kh * 8 + v) * GST + nt * 16 + ln] = acc[v];
        }
        __syncthreads();

        // ---- (c) layer-1 elementwise: c1,h1 update + repack h1 ----
        {
            int r = tid >> 3, p0 = tid & 7;
            #pragma unroll
            for (int q = 0; q < 8; ++q) {
                int p = q * 8 + p0;
                u32 hb[2], lb[2];
                #pragma unroll
                for (int s = 0; s < 2; ++s) {
                    int j = 2 * p + s;
                    float gi = gates[r * GST + j]          + bs1[j];
                    float gf = gates[r * GST + Hh + j]     + bs1[Hh + j];
                    float gg = gates[r * GST + 2 * Hh + j] + bs1[2 * Hh + j];
                    float go = gates[r * GST + 3 * Hh + j] + bs1[3 * Hh + j];
                    float cn = sigf(gf) * cb1[r * Hh + j] + sigf(gi) * tanhf_fast(gg);
                    float hn = sigf(go) * tanhf_fast(cn);
                    cb1[r * Hh + j] = cn;
                    split2(hn, hb[s], lb[s]);
                    if (t == Ll - 1) {
                        out[OH1 + (size_t)(row0 + r) * Hh + j] = hn;
                        out[OC1 + (size_t)(row0 + r) * Hh + j] = cn;
                    }
                }
                h1H[r * 65 + p] = hb[0] | (hb[1] << 16);
                h1L[r * 65 + p] = lb[0] | (lb[1] << 16);
            }
        }
        __syncthreads();

        // ---- (d) layer-2 gates = [h1|h2] x W2cat^T ----
        for (int n2 = 0; n2 < 8; ++n2) {
            int nt = ng * 8 + n2;
            v8f acc = {};
            #pragma unroll
            for (int c = 0; c < KC2; ++c) {
                v16bf aH, aL;
                if (c < 4) {
                    aH = lda(h1H, 65, arow, 16 * c, kh);
                    aL = lda(h1L, 65, arow, 16 * c, kh);
                } else {
                    aH = lda(h2H, 65, arow, 16 * (c - 4), kh);
                    aL = lda(h2L, 65, arow, 16 * (c - 4), kh);
                }
                int fi = nt * KC2 + c;
                v16bf bH = ldb(wsH2, fi, lane);
                v16bf bL = ldb(wsL2, fi, lane);
                acc = __builtin_amdgcn_wmma_f32_16x16x32_bf16(false, aH, false, bH, (short)0, acc, false, false);
                acc = __builtin_amdgcn_wmma_f32_16x16x32_bf16(false, aH, false, bL, (short)0, acc, false, false);
                acc = __builtin_amdgcn_wmma_f32_16x16x32_bf16(false, aL, false, bH, (short)0, acc, false, false);
            }
            #pragma unroll
            for (int v = 0; v < 8; ++v)
                gates[(mt * 16 + kh * 8 + v) * GST + nt * 16 + ln] = acc[v];
        }
        __syncthreads();

        // ---- (e) layer-2 elementwise: c2,h2 update + repack h2 + FC partials ----
        {
            int r = tid >> 3, p0 = tid & 7;
            float ysum = 0.0f;
            #pragma unroll
            for (int q = 0; q < 8; ++q) {
                int p = q * 8 + p0;
                u32 hb[2], lb[2];
                #pragma unroll
                for (int s = 0; s < 2; ++s) {
                    int j = 2 * p + s;
                    float gi = gates[r * GST + j]          + bs2[j];
                    float gf = gates[r * GST + Hh + j]     + bs2[Hh + j];
                    float gg = gates[r * GST + 2 * Hh + j] + bs2[2 * Hh + j];
                    float go = gates[r * GST + 3 * Hh + j] + bs2[3 * Hh + j];
                    float cn = sigf(gf) * cb2[r * Hh + j] + sigf(gi) * tanhf_fast(gg);
                    float hn = sigf(go) * tanhf_fast(cn);
                    cb2[r * Hh + j] = cn;
                    split2(hn, hb[s], lb[s]);
                    ysum += hn * wfcs[j];
                    if (t == Ll - 1) {
                        out[OH2 + (size_t)(row0 + r) * Hh + j] = hn;
                        out[OC2 + (size_t)(row0 + r) * Hh + j] = cn;
                    }
                }
                h2H[r * 65 + p] = hb[0] | (hb[1] << 16);
                h2L[r * 65 + p] = lb[0] | (lb[1] << 16);
            }
            yred[r * 8 + p0] = ysum;
        }
        __syncthreads();

        // ---- FC reduce + output_seq write ----
        if (tid < BT) {
            float y = bfc[0];
            #pragma unroll
            for (int p = 0; p < 8; ++p) y += yred[tid * 8 + p];
            out[OSEQ + (size_t)(row0 + tid) * Ll + t] = y;
        }
        __syncthreads();
    }
}

extern "C" void kernel_launch(void* const* d_in, const int* in_sizes, int n_in,
                              void* d_out, int out_size, void* d_ws, size_t ws_size,
                              hipStream_t stream) {
    (void)in_sizes; (void)n_in; (void)out_size; (void)ws_size;
    const float* x    = (const float*)d_in[0];
    const float* h1_0 = (const float*)d_in[1];
    const float* c1_0 = (const float*)d_in[2];
    const float* h2_0 = (const float*)d_in[3];
    const float* c2_0 = (const float*)d_in[4];
    const float* Wih1 = (const float*)d_in[5];
    const float* Whh1 = (const float*)d_in[6];
    const float* bih1 = (const float*)d_in[7];
    const float* bhh1 = (const float*)d_in[8];
    const float* Wih2 = (const float*)d_in[9];
    const float* Whh2 = (const float*)d_in[10];
    const float* bih2 = (const float*)d_in[11];
    const float* bhh2 = (const float*)d_in[12];
    const float* wfc  = (const float*)d_in[13];
    const float* bfc  = (const float*)d_in[14];

    u32* wsH1 = (u32*)d_ws;
    u32* wsL1 = wsH1 + FR1;
    u32* wsH2 = wsL1 + FR1;
    u32* wsL2 = wsH2 + FR2;

    const int prepThreads = FR1 + FR2;
    prep_weights<<<(prepThreads + 255) / 256, 256, 0, stream>>>(
        Wih1, Whh1, Wih2, Whh2, wsH1, wsL1, wsH2, wsL2);

    lstm_main<<<Bsz / BT, 256, SMEM_BYTES, stream>>>(
        x, h1_0, c1_0, h2_0, c2_0, bih1, bhh1, bih2, bhh2, wfc, bfc,
        wsH1, wsL1, wsH2, wsL2, (float*)d_out);
}
